// Memory_70978629533986
// MI455X (gfx1250) — matrix operand order repaired and checked by reading, not compiled, not run on hardware
//
#include <hip/hip_runtime.h>
#include <hip/hip_bf16.h>
#include <math.h>

// ---------------------------------------------------------------------------
// MI455X (gfx1250) implementation.
//   * compute-bound problem (~500 FLOP/byte) -> use v_wmma_f32_16x16x32_bf16
//   * fp32 accuracy recovered via bf16 hi/lo split (3 WMMAs per tile-K step)
//   * wave32, 8 waves / 256-thread block; wave tile 64x32 (4x2 WMMA subtiles)
//     -> 24 WMMAs per K=32 step against ~112 conversion VALU ops, so the
//     fp32->bf16 split of the B operand co-executes under the matrix pipe.
// ---------------------------------------------------------------------------

#define B_ROWS   1024
#define DIM      2048
#define CLS      2048
#define PN_      8192
#define PPOS     4
#define NUM_CAMS 6
#define BG_KNN_  50
#define EPSV     0.1f
#define INV_TEMP 20.0f   // 1 / 0.05

#define MI_T 4           // 16-row subtiles per wave (M)
#define NI_T 2           // 16-col subtiles per wave (N)

typedef __bf16 bf16_t;
typedef __attribute__((ext_vector_type(16))) __bf16        v16bf;
typedef __attribute__((ext_vector_type(8)))  float         v8f;
typedef __attribute__((ext_vector_type(4)))  unsigned int  uv4;
typedef __attribute__((ext_vector_type(4)))  float         fv4;

static __device__ __forceinline__ bf16_t bits_to_bf16(unsigned short s) {
    union { unsigned short u; bf16_t b; } cv; cv.u = s; return cv.b;
}

// Split fp32 -> bf16 hi (truncate) + bf16 lo (truncate of remainder).
static __device__ __forceinline__ void split_bf16(float x, bf16_t& hi, bf16_t& lo) {
    unsigned u  = __float_as_uint(x);
    unsigned hu = u & 0xFFFF0000u;
    float    r  = x - __uint_as_float(hu);
    unsigned lu = __float_as_uint(r);
    hi = bits_to_bf16((unsigned short)(hu >> 16));
    lo = bits_to_bf16((unsigned short)(lu >> 16));
}

// --------------------------- prepass: split A ------------------------------
__global__ __launch_bounds__(256)
void split_inputs_kernel(const float* __restrict__ x,
                         unsigned short* __restrict__ hi,
                         unsigned short* __restrict__ lo, int n) {
    int i = blockIdx.x * blockDim.x + threadIdx.x;
    if (i < n) {
        float v = x[i];
        unsigned u  = __float_as_uint(v);
        unsigned hu = u & 0xFFFF0000u;
        float    r  = v - __uint_as_float(hu);
        unsigned lu = __float_as_uint(r);
        hi[i] = (unsigned short)(hu >> 16);
        lo[i] = (unsigned short)(lu >> 16);
    }
}

// --------------------------- WMMA GEMM -------------------------------------
// out[b, n] = (sum_k A[b,k] * Bm[n,k]) * INV_TEMP
// A given pre-split (hi/lo bf16, row-major [B_ROWS, DIM]);
// Bm is fp32 row-major [N, DIM], split on the fly (same bytes as bf16 hi+lo,
// so zero bandwidth penalty; L2 (192 MB) holds the full working set).
// Block: 256 threads = 8 waves arranged 2x4; block tile 128 rows x 128 cols;
// each wave owns a 64x32 tile = 4x2 WMMA subtiles.
__global__ __launch_bounds__(256)
void gemm_bf16x2_kernel(const unsigned short* __restrict__ Ahi16,
                        const unsigned short* __restrict__ Alo16,
                        const float* __restrict__ Bm,
                        float* __restrict__ out, int N) {
    const bf16_t* Ahi = reinterpret_cast<const bf16_t*>(Ahi16);
    const bf16_t* Alo = reinterpret_cast<const bf16_t*>(Alo16);

    const int lane = threadIdx.x & 31;
    const int wave = threadIdx.x >> 5;
    const int wrow = wave >> 2;          // 0..1
    const int wcol = wave & 3;           // 0..3
    const int m_base = blockIdx.y * (MI_T * 16 * 2) + wrow * (MI_T * 16);
    const int n_base = blockIdx.x * (NI_T * 16 * 4) + wcol * (NI_T * 16);
    const int half = lane >> 4;          // 0: lanes 0-15, 1: lanes 16-31
    const int l15  = lane & 15;

    v8f acc[MI_T][NI_T] = {};

    for (int k = 0; k < DIM; k += 32) {
        // ---- A operand: 16-bit A 16x32 layout.
        // lanes 0-15 : elems 0..7 = K k..k+7,    elems 8..15 = K k+16..k+23
        // lanes 16-31: elems 0..7 = K k+8..k+15, elems 8..15 = K k+24..k+31
        v16bf ahi[MI_T], alo[MI_T];
#pragma unroll
        for (int mi = 0; mi < MI_T; ++mi) {
            size_t row = (size_t)(m_base + mi * 16 + l15);
            const bf16_t* pa = Ahi + row * DIM + k + half * 8;
            const bf16_t* pl = Alo + row * DIM + k + half * 8;
            union { v16bf v; uv4 q[2]; } th, tl;
            th.q[0] = *(const uv4*)(pa);
            th.q[1] = *(const uv4*)(pa + 16);
            tl.q[0] = *(const uv4*)(pl);
            tl.q[1] = *(const uv4*)(pl + 16);
            ahi[mi] = th.v; alo[mi] = tl.v;
        }
        // ---- B operand: 16-bit B 32x16 layout (B^T column = center row,
        // contiguous in memory). lane<16: col=lane, K k..k+7,k+16..k+23; lane>=16:
        // col=lane-16, K k+8..k+15,k+24..k+31. Convert fp32 -> hi/lo bf16 in-register.
        v16bf bhi[NI_T], blo[NI_T];
#pragma unroll
        for (int ni = 0; ni < NI_T; ++ni) {
            size_t col = (size_t)(n_base + ni * 16 + l15);
            const float* pb = Bm + col * DIM + k + half * 8;
            v16bf h, l;
#pragma unroll
            for (int j = 0; j < 16; j += 4) {
                fv4 f = *(const fv4*)(pb + j + (j & 8));
#pragma unroll
                for (int e = 0; e < 4; ++e) {
                    bf16_t hb, lb;
                    split_bf16(f[e], hb, lb);
                    h[j + e] = hb;
                    l[j + e] = lb;
                }
            }
            bhi[ni] = h; blo[ni] = l;
        }
        // ---- 24 WMMAs: compensated product hi*hi + hi*lo + lo*hi
#pragma unroll
        for (int mi = 0; mi < MI_T; ++mi)
#pragma unroll
            for (int ni = 0; ni < NI_T; ++ni) {
                acc[mi][ni] = __builtin_amdgcn_wmma_f32_16x16x32_bf16(
                    false, ahi[mi], false, bhi[ni], (short)0, acc[mi][ni], false, false);
                acc[mi][ni] = __builtin_amdgcn_wmma_f32_16x16x32_bf16(
                    false, ahi[mi], false, blo[ni], (short)0, acc[mi][ni], false, false);
                acc[mi][ni] = __builtin_amdgcn_wmma_f32_16x16x32_bf16(
                    false, alo[mi], false, bhi[ni], (short)0, acc[mi][ni], false, false);
            }
    }

    // ---- store: C/D 16x16 f32 layout: VGPR r -> M=r (lanes 0-15) / r+8.
#pragma unroll
    for (int mi = 0; mi < MI_T; ++mi)
#pragma unroll
        for (int ni = 0; ni < NI_T; ++ni) {
            int col   = n_base + ni * 16 + l15;
            int rbase = m_base + mi * 16 + half * 8;
#pragma unroll
            for (int r = 0; r < 8; ++r)
                out[(size_t)(rbase + r) * N + col] = acc[mi][ni][r] * INV_TEMP;
        }
}

// ---------------- row epilogue: CE-label-smooth + soft entropy -------------
__global__ __launch_bounds__(256)
void row_softmax_kernel(const float* __restrict__ cluster_sim,
                        const float* __restrict__ proxy_sim2,
                        const int* __restrict__ indexes,
                        const int* __restrict__ labels,
                        float* __restrict__ cel, float* __restrict__ sel) {
    __shared__ float red[256];
    __shared__ float red2[256];
    __shared__ float s_cst;
    const int b = blockIdx.x, tid = threadIdx.x;
    const int t = labels[indexes[b]];
    const float* cs = cluster_sim + (size_t)b * CLS;
    const float* p2 = proxy_sim2 + (size_t)b * CLS;

    float m1 = -1e30f, m2 = -1e30f;
    for (int i = tid; i < CLS; i += 256) {
        float c = cs[i];
        m1 = fmaxf(m1, c);
        m2 = fmaxf(m2, p2[i]);
        if (i == t) s_cst = c;
    }
    red[tid] = m1; red2[tid] = m2; __syncthreads();
    for (int s = 128; s > 0; s >>= 1) {
        if (tid < s) {
            red[tid]  = fmaxf(red[tid],  red[tid + s]);
            red2[tid] = fmaxf(red2[tid], red2[tid + s]);
        }
        __syncthreads();
    }
    m1 = red[0]; m2 = red2[0]; __syncthreads();

    float s1 = 0.f, s2 = 0.f, sumcs = 0.f, dot = 0.f;
    for (int i = tid; i < CLS; i += 256) {
        float c = cs[i], q = p2[i];
        float e1 = __expf(c - m1);
        s1 += e1;
        s2 += __expf(q - m2);
        sumcs += c;
        dot += e1 * q;
    }
    red[tid] = s1; red2[tid] = s2; __syncthreads();
    for (int s = 128; s > 0; s >>= 1) {
        if (tid < s) { red[tid] += red[tid + s]; red2[tid] += red2[tid + s]; }
        __syncthreads();
    }
    s1 = red[0]; s2 = red2[0]; __syncthreads();
    red[tid] = sumcs; red2[tid] = dot; __syncthreads();
    for (int s = 128; s > 0; s >>= 1) {
        if (tid < s) { red[tid] += red[tid + s]; red2[tid] += red2[tid + s]; }
        __syncthreads();
    }
    if (tid == 0) {
        sumcs = red[0]; dot = red2[0];
        float logZ1 = m1 + __logf(s1);
        float logZ2 = m2 + __logf(s2);
        cel[b] = -((1.0f - EPSV) * (s_cst - logZ1) +
                   (EPSV / (float)CLS) * (sumcs - (float)CLS * logZ1));
        sel[b] = logZ2 - dot / s1;   // sum(soft_t)==1
    }
}

// ---------------- top-(P+50) proxy loss ------------------------------------
// Positives are forced to 1e4 in the reference, so the selected set is
// exactly {4 positives} U {top-50 of the rest}; with a uniform 1/P target:
//   per_loss = logsumexp(set) - mean(positive values)
__global__ __launch_bounds__(256)
void topk_loss_kernel(const float* __restrict__ proxy_sim,
                      const int* __restrict__ indexes,
                      const int* __restrict__ labels,
                      const int* __restrict__ c2p,
                      float* __restrict__ perloss) {
    __shared__ float sim[PN_];
    __shared__ float rv[256];
    __shared__ int   ri[256];
    __shared__ float s_posv[PPOS];
    __shared__ float s_M, s_sexp;
    const int b = blockIdx.x, tid = threadIdx.x;
    const float* row = proxy_sim + (size_t)b * PN_;

    for (int i = tid; i < PN_; i += 256) sim[i] = row[i];
    __syncthreads();

    const int t = labels[indexes[b]];
    if (tid < PPOS) {
        int p = c2p[t * PPOS + tid];
        s_posv[tid] = sim[p];
        sim[p] = -1e30f;                 // exclude positives from top-50 scan
    }
    __syncthreads();

    for (int it = 0; it < BG_KNN_; ++it) {
        float bv = -1e30f; int bi = 0;
        for (int i = tid; i < PN_; i += 256) {
            float v = sim[i];
            if (v > bv) { bv = v; bi = i; }
        }
        rv[tid] = bv; ri[tid] = bi; __syncthreads();
        for (int s = 128; s > 0; s >>= 1) {
            if (tid < s && rv[tid + s] > rv[tid]) {
                rv[tid] = rv[tid + s]; ri[tid] = ri[tid + s];
            }
            __syncthreads();
        }
        if (tid == 0) {
            float v = rv[0];
            if (it == 0) {
                float M = v;
                for (int j = 0; j < PPOS; ++j) M = fmaxf(M, s_posv[j]);
                float se = 0.f;
                for (int j = 0; j < PPOS; ++j) se += __expf(s_posv[j] - M);
                s_M = M; s_sexp = se;
            }
            s_sexp += __expf(v - s_M);
            sim[ri[0]] = -1e30f;         // extract
        }
        __syncthreads();
    }
    if (tid == 0) {
        float psum = s_posv[0] + s_posv[1] + s_posv[2] + s_posv[3];
        perloss[b] = (s_M + __logf(s_sexp)) - (1.0f / PPOS) * psum;
    }
}

// ---------------- final reduction ------------------------------------------
__global__ __launch_bounds__(256)
void final_reduce_kernel(const float* __restrict__ cel,
                         const float* __restrict__ sel,
                         const float* __restrict__ perloss,
                         const int* __restrict__ cams,
                         float* __restrict__ out) {
    __shared__ int   icnt[NUM_CAMS];
    __shared__ float red[256];
    const int tid = threadIdx.x;
    if (tid < NUM_CAMS) icnt[tid] = 0;
    __syncthreads();
    for (int b = tid; b < B_ROWS; b += 256) atomicAdd(&icnt[cams[b]], 1);
    __syncthreads();
    float acc = 0.f;
    for (int b = tid; b < B_ROWS; b += 256) {
        acc += (cel[b] + 10.0f * sel[b]) * (1.0f / (float)B_ROWS);
        acc += perloss[b] / (float)icnt[cams[b]];
    }
    red[tid] = acc; __syncthreads();
    for (int s = 128; s > 0; s >>= 1) {
        if (tid < s) red[tid] += red[tid + s];
        __syncthreads();
    }
    if (tid == 0) out[0] = red[0];
}

// ---------------------------------------------------------------------------
extern "C" void kernel_launch(void* const* d_in, const int* in_sizes, int n_in,
                              void* d_out, int out_size, void* d_ws, size_t ws_size,
                              hipStream_t stream) {
    const float* inputs          = (const float*)d_in[0];
    const int*   indexes         = (const int*)  d_in[1];
    const int*   cams            = (const int*)  d_in[2];
    const int*   labels          = (const int*)  d_in[3];
    const float* cluster_centers = (const float*)d_in[4];
    const float* proxy_centers   = (const float*)d_in[5];
    const float* proxy_centers2  = (const float*)d_in[6];
    const int*   c2p             = (const int*)  d_in[7];
    float* out = (float*)d_out;

    char* ws = (char*)d_ws;
    size_t off = 0;
    auto alloc = [&](size_t bytes) -> void* {
        void* p = ws + off;
        off = (off + bytes + 255) & ~(size_t)255;
        return p;
    };
    unsigned short* Ahi = (unsigned short*)alloc((size_t)B_ROWS * DIM * 2);  // 4 MB
    unsigned short* Alo = (unsigned short*)alloc((size_t)B_ROWS * DIM * 2);  // 4 MB
    float* cluster_sim  = (float*)alloc((size_t)B_ROWS * CLS * 4);           // 8 MB
    float* proxy_sim    = (float*)alloc((size_t)B_ROWS * PN_ * 4);           // 32 MB
    float* proxy_sim2   = (float*)alloc((size_t)B_ROWS * CLS * 4);           // 8 MB
    float* cel          = (float*)alloc(B_ROWS * 4);
    float* sel          = (float*)alloc(B_ROWS * 4);
    float* perloss      = (float*)alloc(B_ROWS * 4);

    // 1) split inputs into bf16 hi/lo
    {
        int n = B_ROWS * DIM;
        split_inputs_kernel<<<(n + 255) / 256, 256, 0, stream>>>(inputs, Ahi, Alo, n);
    }
    // 2) three WMMA GEMMs (scaled by 1/TEMP); block tile 128x128
    gemm_bf16x2_kernel<<<dim3(CLS / 128, B_ROWS / 128), 256, 0, stream>>>(
        Ahi, Alo, cluster_centers, cluster_sim, CLS);
    gemm_bf16x2_kernel<<<dim3(PN_ / 128, B_ROWS / 128), 256, 0, stream>>>(
        Ahi, Alo, proxy_centers, proxy_sim, PN_);
    gemm_bf16x2_kernel<<<dim3(CLS / 128, B_ROWS / 128), 256, 0, stream>>>(
        Ahi, Alo, proxy_centers2, proxy_sim2, CLS);
    // 3) per-row epilogues
    row_softmax_kernel<<<B_ROWS, 256, 0, stream>>>(
        cluster_sim, proxy_sim2, indexes, labels, cel, sel);
    topk_loss_kernel<<<B_ROWS, 256, 0, stream>>>(
        proxy_sim, indexes, labels, c2p, perloss);
    // 4) scalar loss
    final_reduce_kernel<<<1, 256, 0, stream>>>(cel, sel, perloss, cams, out);
}
